// GraphSAGE_75033078661562
// MI455X (gfx1250) — matrix operand order repaired and checked by reading, not compiled, not run on hardware
//
#include <hip/hip_runtime.h>
#include <hip/hip_bf16.h>

typedef __attribute__((ext_vector_type(16))) _Float16 v16h;
typedef __attribute__((ext_vector_type(8)))  float    v8f;

#define D 128

// ---------------------------------------------------------------------------
// Zero a float buffer (count4 = number of float4 elements)
// ---------------------------------------------------------------------------
__global__ void zero_f32(float* __restrict__ p, int count4) {
    int i = blockIdx.x * blockDim.x + threadIdx.x;
    if (i < count4) {
        ((float4*)p)[i] = make_float4(0.f, 0.f, 0.f, 0.f);
    }
}

// ---------------------------------------------------------------------------
// Weight prep: W[k][n] f32 row-major  ->  WT[n][k] f16 (column-contiguous in K)
// 128x128, grid 64x256
// ---------------------------------------------------------------------------
__global__ void transpose_f16(const float* __restrict__ w, _Float16* __restrict__ wt) {
    int t  = blockIdx.x * blockDim.x + threadIdx.x;   // 0..16383
    int nn = t >> 7;
    int kk = t & 127;
    wt[nn * D + kk] = (_Float16)w[kk * D + nn];
}

// ---------------------------------------------------------------------------
// Degree: one thread per edge, float atomic into deg[dst]
// ---------------------------------------------------------------------------
__global__ void degree_kernel(const long long* __restrict__ ei, float* __restrict__ deg, int E) {
    int e = blockIdx.x * blockDim.x + threadIdx.x;
    if (e < E) {
        int d = (int)ei[(size_t)E + e];
        atomicAdd(&deg[d], 1.0f);
    }
}

// ---------------------------------------------------------------------------
// Scatter-add: one wave per edge; lane l handles features [4l, 4l+4)
// Gather is a coalesced 512B row read; atomics land in L2 (agg fits in 192MB L2)
// ---------------------------------------------------------------------------
__global__ void scatter_kernel(const float* __restrict__ feat, const long long* __restrict__ ei,
                               float* __restrict__ agg, int E) {
    int gid  = blockIdx.x * blockDim.x + threadIdx.x;
    int e    = gid >> 5;
    int lane = gid & 31;
    if (e >= E) return;
    int s = (int)ei[e];
    int d = (int)ei[(size_t)E + e];
    const float4 v = *(const float4*)(feat + (size_t)s * D + lane * 4);
    float* a = agg + (size_t)d * D + lane * 4;
    atomicAdd(a + 0, v.x);
    atomicAdd(a + 1, v.y);
    atomicAdd(a + 2, v.z);
    atomicAdd(a + 3, v.w);
}

// ---------------------------------------------------------------------------
// Fused SAGE layer:  out = [relu]( (agg/max(deg,1)) @ Wl + b + feat @ Wr )
// 256 threads = 8 waves; each wave owns a 16-row node tile x all 128 cols.
// Weights (two 128x128 f16 matrices, WT layout [n][k]) staged in 64KB LDS.
// v_wmma_f32_16x16x32_f16, f32 accumulate, 64 WMMA per wave-tile.
// ---------------------------------------------------------------------------
__global__ __launch_bounds__(256) void sage_layer(
    const float*    __restrict__ feat,   // [n][128] f32
    const float*    __restrict__ agg,    // [n][128] f32
    const float*    __restrict__ deg,    // [n] f32
    const _Float16* __restrict__ wt,     // [WTl | WTr], each 128*128, layout [n][k]
    const float*    __restrict__ bias,   // [128]
    float*          __restrict__ out,    // [n][128]
    int n, int do_relu)
{
    __shared__ __align__(32) _Float16 sW[2 * D * D];   // 64 KB

    // cooperative 64KB copy of both weight matrices into LDS
    {
        const uint4* gsrc = (const uint4*)wt;
        uint4*       ldst = (uint4*)sW;
#pragma unroll
        for (int i = 0; i < 16; ++i)
            ldst[threadIdx.x + i * 256] = gsrc[threadIdx.x + i * 256];
    }
    __syncthreads();

    const int lane = threadIdx.x & 31;
    const int wave = threadIdx.x >> 5;
    const int row0 = blockIdx.x * 128 + wave * 16;
    if (row0 >= n) return;

    const int m    = lane & 15;       // A row within tile / B+D column within tile
    const int half = lane >> 4;       // lane group
    int row = row0 + m;
    if (row >= n) row = n - 1;        // clamped load; stores guarded below

    const float rdeg = 1.0f / fmaxf(deg[row], 1.0f);

    const float* fbase = feat + (size_t)row * D;
    const float* abase = agg  + (size_t)row * D;

    const _Float16* wl = sW;
    const _Float16* wr = sW + D * D;
    const int kh = half * 16;         // B fragment K-offset for this lane group

    v8f acc[8];
#pragma unroll
    for (int ct = 0; ct < 8; ++ct) {
        v8f z = {};
        acc[ct] = z;
    }

#pragma unroll
    for (int s = 0; s < 4; ++s) {
        const int k0 = s * 32;
        // Build A fragments in contiguous K order (matching B):
        //   lane<16 : elems 0-15 -> K k0+0..15
        //   lane>=16: elems 0-15 -> K k0+16..31
        const float* pf = fbase + k0 + half * 16;
        const float* pa = abase + k0 + half * 16;
        v16h ax, am;
#pragma unroll
        for (int i = 0; i < 8; ++i) {
            ax[i]     = (_Float16)pf[i];
            ax[i + 8] = (_Float16)pf[i + 8];
            am[i]     = (_Float16)(pa[i] * rdeg);
            am[i + 8] = (_Float16)(pa[i + 8] * rdeg);
        }
#pragma unroll
        for (int ct = 0; ct < 8; ++ct) {
            // B fragment: lane holds 16 contiguous K halves of column ct*16+m
            const v16h bl = *(const v16h*)(wl + (ct * 16 + m) * D + k0 + kh);
            acc[ct] = __builtin_amdgcn_wmma_f32_16x16x32_f16(
                false, am, false, bl, (short)0, acc[ct], false, false);
            const v16h br = *(const v16h*)(wr + (ct * 16 + m) * D + k0 + kh);
            acc[ct] = __builtin_amdgcn_wmma_f32_16x16x32_f16(
                false, ax, false, br, (short)0, acc[ct], false, false);
        }
    }

    // Epilogue: D layout -> VGPR v: lanes 0-15 row v, lanes 16-31 row 8+v; col = lane&15
#pragma unroll
    for (int ct = 0; ct < 8; ++ct) {
        const int col  = ct * 16 + m;
        const float bv = bias[col];
#pragma unroll
        for (int v = 0; v < 8; ++v) {
            const int rr = row0 + half * 8 + v;
            if (rr < n) {
                float val = acc[ct][v] + bv;
                if (do_relu) val = fmaxf(val, 0.0f);
                out[(size_t)rr * D + col] = val;
            }
        }
    }
}

// ---------------------------------------------------------------------------
extern "C" void kernel_launch(void* const* d_in, const int* in_sizes, int n_in,
                              void* d_out, int out_size, void* d_ws, size_t ws_size,
                              hipStream_t stream) {
    const float*     x   = (const float*)d_in[0];
    const long long* ei  = (const long long*)d_in[1];
    const float*     W1l = (const float*)d_in[2];
    const float*     b1  = (const float*)d_in[3];
    const float*     W1r = (const float*)d_in[4];
    const float*     W2l = (const float*)d_in[5];
    const float*     b2  = (const float*)d_in[6];
    const float*     W2r = (const float*)d_in[7];
    float*           out = (float*)d_out;

    const int n = in_sizes[0] / D;      // 100000
    const int E = in_sizes[1] / 2;      // 1600000

    // Workspace layout
    char* ws = (char*)d_ws;
    _Float16* WT  = (_Float16*)ws;                             // 4 * 128*128 f16 = 128 KB
    size_t off = 4 * (size_t)D * D * sizeof(_Float16);
    float* deg = (float*)(ws + off);                           // n f32
    off += ((size_t)n * sizeof(float) + 511) & ~(size_t)511;
    float* agg = (float*)(ws + off);                           // n*128 f32
    off += (size_t)n * D * sizeof(float);
    float* h   = (float*)(ws + off);                           // n*128 f32

    // --- weight prep: f32 [k][n] -> f16 [n][k], order W1l W1r W2l W2r ---
    transpose_f16<<<64, 256, 0, stream>>>(W1l, WT + 0 * D * D);
    transpose_f16<<<64, 256, 0, stream>>>(W1r, WT + 1 * D * D);
    transpose_f16<<<64, 256, 0, stream>>>(W2l, WT + 2 * D * D);
    transpose_f16<<<64, 256, 0, stream>>>(W2r, WT + 3 * D * D);

    // --- degrees (graph is identical for both layers) ---
    zero_f32<<<((n / 4) + 255) / 256, 256, 0, stream>>>(deg, n / 4);
    degree_kernel<<<(E + 255) / 256, 256, 0, stream>>>(ei, deg, E);

    const int agg4       = n * D / 4;
    const int zeroBlocks = (agg4 + 255) / 256;
    const int scatBlocks = (E * 32 + 255) / 256;    // one wave per edge
    const int gemmBlocks = (n + 127) / 128;         // 128 node rows per block

    // --- layer 1: h = relu(mean(x) @ W1l + b1 + x @ W1r) ---
    zero_f32<<<zeroBlocks, 256, 0, stream>>>(agg, agg4);
    scatter_kernel<<<scatBlocks, 256, 0, stream>>>(x, ei, agg, E);
    sage_layer<<<gemmBlocks, 256, 0, stream>>>(x, agg, deg, WT, b1, h, n, 1);

    // --- layer 2: out = mean(h) @ W2l + b2 + h @ W2r ---
    zero_f32<<<zeroBlocks, 256, 0, stream>>>(agg, agg4);
    scatter_kernel<<<scatBlocks, 256, 0, stream>>>(h, ei, agg, E);
    sage_layer<<<gemmBlocks, 256, 0, stream>>>(h, agg, deg, WT + 2 * D * D, b2, out, n, 0);
}
